// DSIB_64939905516327
// MI455X (gfx1250) — matrix operand order repaired and checked by reading, not compiled, not run on hardware
//
#include <hip/hip_runtime.h>
#include <math.h>

#define BATCH 8192
#define NXD   512
#define HIDD  1024
#define EMBD  128

typedef __attribute__((ext_vector_type(16))) __bf16 v16bf;
typedef __attribute__((ext_vector_type(8)))  float  v8f;
typedef __attribute__((ext_vector_type(4)))  unsigned int u32x4;
typedef __attribute__((ext_vector_type(8)))  int i32x8;
typedef __attribute__((ext_vector_type(4)))  int i32x4;

union FragU { v16bf v; float4 f4[2]; };

#if defined(__has_builtin)
#if __has_builtin(__builtin_amdgcn_tensor_load_to_lds) && __has_builtin(__builtin_amdgcn_s_wait_tensorcnt)
#define HAVE_TDM 1
#endif
#endif
#ifndef HAVE_TDM
#define HAVE_TDM 0
#endif

__device__ __forceinline__ unsigned short f2bf(float f) {
    unsigned int u = __float_as_uint(f);
    u += 0x7FFFu + ((u >> 16) & 1u);          // round-to-nearest-even
    return (unsigned short)(u >> 16);
}

#if HAVE_TDM
// Issue one TDM 2-D tile load: tile (tile_d0 x tile_d1) elements (2B each) from a
// row-major tensor with row stride 'stride0' (elements), landing contiguously in LDS.
// D# bitfields per CDNA5 ISA ch.8 (group0: count/lds/global/type, group1: dims).
__device__ __forceinline__ void tdm_load_2d(unsigned int lds_off, const void* gtile,
                                            unsigned int tensor_d0, unsigned int tensor_d1,
                                            unsigned int tile_d0, unsigned int tile_d1,
                                            unsigned int stride0) {
    unsigned long long ga = (unsigned long long)(uintptr_t)gtile;
    u32x4 g0 = {};
    g0[0] = 1u;                                          // count=1 (valid), user mode
    g0[1] = lds_off;                                     // lds_addr (bytes)
    g0[2] = (unsigned int)ga;                            // global_addr[31:0]
    g0[3] = (unsigned int)(ga >> 32) | (2u << 30);       // global_addr[56:32] | type=2
    i32x8 g1 = {};
    g1[0] = (int)(1u << 16);                             // wg_mask=0, data_size=1 (2B)
    g1[1] = (int)((tensor_d0 & 0xFFFFu) << 16);          // tensor_dim0[15:0]
    g1[2] = (int)(((tensor_d0 >> 16) & 0xFFFFu) | ((tensor_d1 & 0xFFFFu) << 16));
    g1[3] = (int)(((tensor_d1 >> 16) & 0xFFFFu) | (tile_d0 << 16));   // tile_dim0
    g1[4] = (int)(tile_d1 & 0xFFFFu);                    // tile_dim1 (tile_dim2=0)
    g1[5] = (int)stride0;                                // tensor_dim0_stride[31:0]
    g1[6] = 0;                                           // stride0 hi | stride1 lo
    g1[7] = 0;
    i32x4 z4 = {};
#if __clang_major__ >= 23
    i32x8 z8 = {};
    __builtin_amdgcn_tensor_load_to_lds(g0, g1, z4, z4, z8, 0);
#else
    __builtin_amdgcn_tensor_load_to_lds(g0, g1, z4, z4, 0);
#endif
}
#endif

// ---------------------------------------------------------------- convert (elementwise)
__global__ void cvt_f32_bf16(const float* __restrict__ in,
                             unsigned short* __restrict__ out, int n) {
    int i = blockIdx.x * blockDim.x + threadIdx.x;
    if (i < n) out[i] = f2bf(in[i]);
}

// ---------------------------------------------------------------- convert + transpose
// in: [K,N] f32 row-major  ->  out: [N,K] bf16 row-major.  LDS-tiled 32x32.
__global__ void cvt_transpose_bf16(const float* __restrict__ in,
                                   unsigned short* __restrict__ out, int K, int N) {
    __shared__ float tile[32][33];
    const int k0 = blockIdx.y * 32, n0 = blockIdx.x * 32;
    const int tx = threadIdx.x & 31, ty = threadIdx.x >> 5;   // 256 thr: 8 rows/pass
#pragma unroll
    for (int r = ty; r < 32; r += 8)
        tile[r][tx] = in[(size_t)(k0 + r) * N + n0 + tx];
    __syncthreads();
#pragma unroll
    for (int r = ty; r < 32; r += 8)
        out[(size_t)(n0 + r) * K + k0 + tx] = f2bf(tile[tx][r]);
}

// ---------------------------------------------------------------- GEMM
// C[M,N](bf16) = act(A[M,K](bf16) @ WT[N,K](bf16)^T + bias)   tile 128x64, K-step 32
// 256 threads = 8 waves; wave w computes rows [w*16, w*16+16) x all 64 cols.
// TDM path: double-buffered LDS tiles filled by tensor_load_to_lds, overlapped
// with the WMMA chain via TENSORcnt (<=2 keeps next tile's DMA in flight).
__global__ void gemm_bias_act(const unsigned short* __restrict__ A,
                              const unsigned short* __restrict__ WT,
                              const float* __restrict__ bias,
                              unsigned short* __restrict__ C,
                              int M, int N, int K, int relu) {
    __shared__ float4 smemV[2 * 768];   // per buf: A 128x32 bf16 (512 f4) + B 64x32 (256 f4)
    const int tid  = threadIdx.x;
    const int w    = tid >> 5;
    const int lane = tid & 31;
    const int g    = lane >> 4;
    const int ln   = lane & 15;
    const int m0   = blockIdx.y * 128;
    const int n0   = blockIdx.x * 64;
    const int T    = K >> 5;

    v8f acc[4];
#pragma unroll
    for (int nt = 0; nt < 4; ++nt) acc[nt] = (v8f){0.f,0.f,0.f,0.f,0.f,0.f,0.f,0.f};

#if HAVE_TDM
    const unsigned int ldsBase = (unsigned int)(uintptr_t)(void*)smemV;
    if (w == 0) {   // wave 0 drives the TDM; EXEC-independent, per-wave TENSORcnt
        tdm_load_2d(ldsBase, A + (size_t)m0 * K, (unsigned)K, (unsigned)M, 32u, 128u, (unsigned)K);
        tdm_load_2d(ldsBase + 512 * 16, WT + (size_t)n0 * K, (unsigned)K, (unsigned)N, 32u, 64u, (unsigned)K);
    }
    for (int t = 0; t < T; ++t) {
        if (w == 0) {
            if (t + 1 < T) {
                const int k0n = (t + 1) * 32;
                const unsigned int lb = ldsBase + ((t + 1) & 1) * 768 * 16;
                tdm_load_2d(lb, A + (size_t)m0 * K + k0n, (unsigned)K, (unsigned)M, 32u, 128u, (unsigned)K);
                tdm_load_2d(lb + 512 * 16, WT + (size_t)n0 * K + k0n, (unsigned)K, (unsigned)N, 32u, 64u, (unsigned)K);
                __builtin_amdgcn_s_wait_tensorcnt(2);   // tile t landed; t+1 in flight
            } else {
                __builtin_amdgcn_s_wait_tensorcnt(0);
            }
        }
        __syncthreads();
        const unsigned short* As = (const unsigned short*)(smemV + 768 * (t & 1));
        const unsigned short* Bs = As + 512 * 8;
#else
    for (int t = 0; t < T; ++t) {
        const int k0 = t * 32;
        __syncthreads();
#pragma unroll
        for (int i = tid; i < 512; i += 256) {          // A tile, coalesced f4
            int row = i >> 2, seg = i & 3;
            smemV[i] = *(const float4*)(A + (size_t)(m0 + row) * K + k0 + seg * 8);
        }
        {                                               // B tile (already n-major)
            int n = tid >> 2, kseg = tid & 3;
            smemV[512 + n * 4 + kseg] = *(const float4*)(WT + (size_t)(n0 + n) * K + k0 + kseg * 8);
        }
        __syncthreads();
        const unsigned short* As = (const unsigned short*)smemV;
        const unsigned short* Bs = As + 512 * 8;
#endif
        // A fragment (16x32 bf16), k = 16*g + s per lane half (contiguous order, same as B)
        FragU a;
        {
            const float4* ap = (const float4*)(As + (w * 16 + ln) * 32);
            a.f4[0] = ap[2 * g];        // K = g*16 .. g*16+7
            a.f4[1] = ap[2 * g + 1];    // K = g*16+8 .. g*16+15
        }
#pragma unroll
        for (int nt = 0; nt < 4; ++nt) {
            FragU b;
            const float4* bp = (const float4*)(Bs + (nt * 16 + ln) * 32);
            b.f4[0] = bp[g * 2];
            b.f4[1] = bp[g * 2 + 1];
            acc[nt] = __builtin_amdgcn_wmma_f32_16x16x32_bf16(
                false, a.v, false, b.v, (short)0, acc[nt], false, false);
        }
        __syncthreads();
    }

    // epilogue: bias + optional relu, bf16 store
#pragma unroll
    for (int nt = 0; nt < 4; ++nt) {
        int col = n0 + nt * 16 + ln;
        float bv = bias[col];
#pragma unroll
        for (int r = 0; r < 8; ++r) {
            int row = m0 + w * 16 + r + g * 8;
            float v = acc[nt][r] + bv;
            if (relu) v = fmaxf(v, 0.0f);
            C[(size_t)row * N + col] = f2bf(v);
        }
    }
}

// ---------------------------------------------------------------- init
__global__ void init_out(float* out) {
    if (threadIdx.x == 0 && blockIdx.x == 0) out[0] = -logf((float)BATCH);
}

// ---------------------------------------------------------------- fused scores + online LSE
__global__ void scores_lse(const unsigned short* __restrict__ ZX,
                           const unsigned short* __restrict__ ZY,
                           float* __restrict__ out) {
    __shared__ float4 XsV[256];          // 16 x 128 bf16
    __shared__ float pm[8][16];
    __shared__ float ps[8][16];
    __shared__ float posS[16];
    unsigned short* Xs = (unsigned short*)XsV;

    const int tid  = threadIdx.x;
    const int w    = tid >> 5;
    const int lane = tid & 31;
    const int g    = lane >> 4;
    const int ln   = lane & 15;
    const int rowBase = blockIdx.x * 16;

    {
        int row = tid >> 4, seg = (tid & 15) * 8;
        *(float4*)(Xs + row * EMBD + seg) =
            *(const float4*)(ZX + (size_t)(rowBase + row) * EMBD + seg);
    }
    __syncthreads();

    FragU a[4];
#pragma unroll
    for (int kt = 0; kt < 4; ++kt) {
        const float4* ap = (const float4*)(Xs + ln * EMBD + kt * 32);
        a[kt].f4[0] = ap[2 * g];
        a[kt].f4[1] = ap[2 * g + 1];
    }

    float run_m[8], run_s[8];
#pragma unroll
    for (int r = 0; r < 8; ++r) { run_m[r] = -INFINITY; run_s[r] = 0.0f; }

    for (int t = 0; t < BATCH / 128; ++t) {
        const int j = t * 128 + w * 16;
        v8f acc = (v8f){0.f,0.f,0.f,0.f,0.f,0.f,0.f,0.f};
#pragma unroll
        for (int kt = 0; kt < 4; ++kt) {
            FragU b;   // B = zY^T tile: column n is zY row (j+n), K-contiguous
            const float4* bp = (const float4*)(ZY + (size_t)(j + ln) * EMBD + kt * 32 + g * 16);
            b.f4[0] = bp[0];
            b.f4[1] = bp[1];
            acc = __builtin_amdgcn_wmma_f32_16x16x32_bf16(
                false, a[kt].v, false, b.v, (short)0, acc, false, false);
        }
#pragma unroll
        for (int r = 0; r < 8; ++r) {
            float s = acc[r];
            int row = rowBase + r + g * 8;
            int col = j + ln;
            if (row == col) posS[row - rowBase] = s;
            float tm = s;
#pragma unroll
            for (int off = 8; off >= 1; off >>= 1) tm = fmaxf(tm, __shfl_xor(tm, off, 16));
            float nm = fmaxf(run_m[r], tm);
            float p = __expf(s - nm);
#pragma unroll
            for (int off = 8; off >= 1; off >>= 1) p += __shfl_xor(p, off, 16);
            run_s[r] = run_s[r] * __expf(run_m[r] - nm) + p;
            run_m[r] = nm;
        }
    }

    if (ln == 0) {
#pragma unroll
        for (int r = 0; r < 8; ++r) {
            pm[w][r + g * 8] = run_m[r];
            ps[w][r + g * 8] = run_s[r];
        }
    }
    __syncthreads();

    if (tid < 16) {
        float M = -INFINITY;
#pragma unroll
        for (int ww = 0; ww < 8; ++ww) M = fmaxf(M, pm[ww][tid]);
        float S = 0.0f;
#pragma unroll
        for (int ww = 0; ww < 8; ++ww) S += ps[ww][tid] * __expf(pm[ww][tid] - M);
        float lse = M + __logf(S);
        float val = posS[tid] - lse;
        atomicAdd(out, -val * (1.0f / (float)BATCH));
    }
}

// ---------------------------------------------------------------- launch
extern "C" void kernel_launch(void* const* d_in, const int* in_sizes, int n_in,
                              void* d_out, int out_size, void* d_ws, size_t ws_size,
                              hipStream_t stream) {
    (void)in_sizes; (void)n_in; (void)out_size; (void)ws_size;
    const float* dataX = (const float*)d_in[0];
    const float* dataY = (const float*)d_in[1];
    const float* Wx0 = (const float*)d_in[2];  const float* bx0 = (const float*)d_in[3];
    const float* Wx1 = (const float*)d_in[4];  const float* bx1 = (const float*)d_in[5];
    const float* Wx2 = (const float*)d_in[6];  const float* bx2 = (const float*)d_in[7];
    const float* Wy0 = (const float*)d_in[8];  const float* by0 = (const float*)d_in[9];
    const float* Wy1 = (const float*)d_in[10]; const float* by1 = (const float*)d_in[11];
    const float* Wy2 = (const float*)d_in[12]; const float* by2 = (const float*)d_in[13];
    float* out = (float*)d_out;

    char* ws = (char*)d_ws;
    size_t off = 0;
    auto take = [&](size_t bytes) { char* p = ws + off; off = (off + bytes + 255) & ~(size_t)255; return p; };
    unsigned short* Xb   = (unsigned short*)take((size_t)BATCH * NXD  * 2);
    unsigned short* Yb   = (unsigned short*)take((size_t)BATCH * NXD  * 2);
    unsigned short* Wx0t = (unsigned short*)take((size_t)NXD  * HIDD * 2);  // [HID,NX]
    unsigned short* Wx1t = (unsigned short*)take((size_t)HIDD * HIDD * 2);  // [HID,HID]
    unsigned short* Wx2t = (unsigned short*)take((size_t)HIDD * EMBD * 2);  // [EMB,HID]
    unsigned short* Wy0t = (unsigned short*)take((size_t)NXD  * HIDD * 2);
    unsigned short* Wy1t = (unsigned short*)take((size_t)HIDD * HIDD * 2);
    unsigned short* Wy2t = (unsigned short*)take((size_t)HIDD * EMBD * 2);
    unsigned short* H0   = (unsigned short*)take((size_t)BATCH * HIDD * 2);
    unsigned short* H1   = (unsigned short*)take((size_t)BATCH * HIDD * 2);
    unsigned short* Zx   = (unsigned short*)take((size_t)BATCH * EMBD * 2);
    unsigned short* Zy   = (unsigned short*)take((size_t)BATCH * EMBD * 2);

    cvt_f32_bf16<<<(BATCH * NXD + 255) / 256, 256, 0, stream>>>(dataX, Xb, BATCH * NXD);
    cvt_f32_bf16<<<(BATCH * NXD + 255) / 256, 256, 0, stream>>>(dataY, Yb, BATCH * NXD);
    // weights: convert + transpose so GEMM B tiles are K-contiguous
    cvt_transpose_bf16<<<dim3(HIDD / 32, NXD / 32), 256, 0, stream>>>(Wx0, Wx0t, NXD, HIDD);
    cvt_transpose_bf16<<<dim3(HIDD / 32, HIDD / 32), 256, 0, stream>>>(Wx1, Wx1t, HIDD, HIDD);
    cvt_transpose_bf16<<<dim3(EMBD / 32, HIDD / 32), 256, 0, stream>>>(Wx2, Wx2t, HIDD, EMBD);
    cvt_transpose_bf16<<<dim3(HIDD / 32, NXD / 32), 256, 0, stream>>>(Wy0, Wy0t, NXD, HIDD);
    cvt_transpose_bf16<<<dim3(HIDD / 32, HIDD / 32), 256, 0, stream>>>(Wy1, Wy1t, HIDD, HIDD);
    cvt_transpose_bf16<<<dim3(EMBD / 32, HIDD / 32), 256, 0, stream>>>(Wy2, Wy2t, HIDD, EMBD);

    dim3 blk(256);
    gemm_bias_act<<<dim3(HIDD / 64, BATCH / 128), blk, 0, stream>>>(Xb, Wx0t, bx0, H0, BATCH, HIDD, NXD, 1);
    gemm_bias_act<<<dim3(HIDD / 64, BATCH / 128), blk, 0, stream>>>(H0, Wx1t, bx1, H1, BATCH, HIDD, HIDD, 1);
    gemm_bias_act<<<dim3(EMBD / 64, BATCH / 128), blk, 0, stream>>>(H1, Wx2t, bx2, Zx, BATCH, EMBD, HIDD, 0);
    gemm_bias_act<<<dim3(HIDD / 64, BATCH / 128), blk, 0, stream>>>(Yb, Wy0t, by0, H0, BATCH, HIDD, NXD, 1);
    gemm_bias_act<<<dim3(HIDD / 64, BATCH / 128), blk, 0, stream>>>(H0, Wy1t, by1, H1, BATCH, HIDD, HIDD, 1);
    gemm_bias_act<<<dim3(EMBD / 64, BATCH / 128), blk, 0, stream>>>(H1, Wy2t, by2, Zy, BATCH, EMBD, HIDD, 0);

    init_out<<<1, 32, 0, stream>>>(out);
    scores_lse<<<BATCH / 16, 256, 0, stream>>>(Zx, Zy, out);
}
